// ManagerStateTracker_29661044146287
// MI455X (gfx1250) — compile-verified, matrix operand order checked by reading, not run on hardware
//
#include <hip/hip_runtime.h>
#include <math.h>

typedef __attribute__((ext_vector_type(16))) _Float16 v16h;
typedef __attribute__((ext_vector_type(8)))  _Float16 v8h;
typedef __attribute__((ext_vector_type(4)))  _Float16 v4h;
typedef __attribute__((ext_vector_type(8)))  float    v8f;
typedef __attribute__((ext_vector_type(4)))  float    v4f;

#define B_      1024
#define P_      64
#define N_      512
#define E_      256
#define F_      64
#define OUTW    12800
#define CONCATW 12544
#define LDA     264   // padded f16 row stride for A tile in LDS (264*2B = 528B -> 4-bank shift/row)

// ---------------------------------------------------------------------------
// K0: W (E,E) f32 row-major  ->  Wt (E,E) f16 stored N-major: Wt[f*E + e] = W[e*E + f]
// Lets the WMMA B-fragment load be contiguous-in-K per lane (lane = N column).
// ---------------------------------------------------------------------------
__global__ void wt_convert_kernel(const float* __restrict__ W, _Float16* __restrict__ Wt) {
    int tid = blockIdx.x * blockDim.x + threadIdx.x;   // 0 .. E*E-1
    int f = tid >> 8;
    int e = tid & (E_ - 1);
    Wt[f * E_ + e] = (_Float16)W[e * E_ + f];
}

// ---------------------------------------------------------------------------
// K1: concat passthrough columns [0, 12544) of the output (coalesced stream)
// ---------------------------------------------------------------------------
__global__ void concat_kernel(const float* __restrict__ feas,     // (B,64)
                              const float* __restrict__ workers,  // (B,128)
                              const float* __restrict__ known,    // (B,4096)
                              const float* __restrict__ unknown,  // (B,4096)
                              const float* __restrict__ differ,   // (B,4096)
                              const float* __restrict__ qaturn,   // (B,64)
                              float* __restrict__ out) {
    long long tid = (long long)blockIdx.x * blockDim.x + threadIdx.x;
    if (tid >= (long long)B_ * CONCATW) return;
    int b = (int)(tid / CONCATW);
    int c = (int)(tid % CONCATW);
    float v;
    if      (c < 64)     v = feas   [b * 64   +  c];
    else if (c < 192)    v = workers[b * 128  + (c - 64)];
    else if (c < 4288)   v = known  [b * 4096 + (c - 192)];
    else if (c < 8384)   v = unknown[b * 4096 + (c - 4288)];
    else if (c < 12480)  v = differ [b * 4096 + (c - 8384)];
    else                 v = qaturn [b * 64   + (c - 12480)];
    out[(long long)b * OUTW + c] = v;
}

// ---------------------------------------------------------------------------
// K2: per-batch gather -> f16 WMMA GEMM (64x256 @ 256x256) -> +bias, tanh,
//     mean over P -> out[b, 12544:12800]
// Block = 256 threads = 8 waves. Wave w owns N-tiles {2w, 2w+1}; each N-tile
// runs 4 M-tile accumulators over 8 K-steps of v_wmma_f32_16x16x32_f16.
// ---------------------------------------------------------------------------
__global__ __launch_bounds__(256) void gemm_agg_kernel(
    const int*      __restrict__ nodes,   // (B,P) int32
    const float*    __restrict__ embed,   // (B,N,E) f32
    const _Float16* __restrict__ Wt,      // (E,E) f16, N-major
    const float*    __restrict__ bias,    // (E,) f32
    float*          __restrict__ out) {

    __shared__ _Float16 Al[P_ * LDA];     // 33792 B
    __shared__ float    sAgg[E_];         // 1024 B

    const int b = blockIdx.x;
    const int t = threadIdx.x;

    sAgg[t] = 0.0f;

    // ---- gather + f32->f16 convert into LDS: 4 threads per row, 64 cols each
    {
        const int row  = t >> 2;          // 0..63
        const int part = t & 3;           // 0..3
        const int node = nodes[b * P_ + row];
        const float*    src = embed + ((long long)b * N_ + node) * E_ + part * 64;
        _Float16*       dst = &Al[row * LDA + part * 64];
        #pragma unroll
        for (int i = 0; i < 16; ++i) {
            v4f x = *(const v4f*)(src + i * 4);
            v4h h; h.x = (_Float16)x.x; h.y = (_Float16)x.y;
                   h.z = (_Float16)x.z; h.w = (_Float16)x.w;
            *(v4h*)(dst + i * 4) = h;
        }
    }
    __syncthreads();

    // ---- WMMA GEMM
    const int wave = t >> 5;
    const int lane = t & 31;
    const int l    = lane & 15;           // M row in tile (A) / N col in tile (B, C/D)
    const int hi8  = lane >> 4;           // K-half select per ISA A/B layout

    #pragma unroll
    for (int nti = 0; nti < 2; ++nti) {
        const int nt   = wave * 2 + nti;
        const int ncol = nt * 16 + l;
        const _Float16* wtb = Wt + ncol * E_;

        v8f acc[4];
        #pragma unroll
        for (int mt = 0; mt < 4; ++mt) acc[mt] = (v8f){0,0,0,0,0,0,0,0};

        #pragma unroll
        for (int kt = 0; kt < 8; ++kt) {
            const int kofs = kt * 32 + hi8 * 8;
            // B fragment: column ncol of W, halves K[kofs..+7] and K[kofs+16..+23]
            v8h blo = *(const v8h*)(wtb + kofs);
            v8h bhi = *(const v8h*)(wtb + kofs + 16);
            v16h bf = __builtin_shufflevector(blo, bhi,
                       0,1,2,3,4,5,6,7,8,9,10,11,12,13,14,15);
            #pragma unroll
            for (int mt = 0; mt < 4; ++mt) {
                const _Float16* ab = &Al[(mt * 16 + l) * LDA + kofs];
                v8h alo = *(const v8h*)ab;          // ds_load_b128
                v8h ahi = *(const v8h*)(ab + 16);   // ds_load_b128
                v16h af = __builtin_shufflevector(alo, ahi,
                           0,1,2,3,4,5,6,7,8,9,10,11,12,13,14,15);
                acc[mt] = __builtin_amdgcn_wmma_f32_16x16x32_f16(
                    /*neg_a=*/false, af, /*neg_b=*/false, bf,
                    /*c_mod=*/(short)0, acc[mt],
                    /*reuse_a=*/false, /*reuse_b=*/false);
            }
        }

        // ---- epilogue: +bias, tanh, partial sum over this lane's 8 M rows x 4 M-tiles
        const float bc = bias[ncol];
        float s = 0.0f;
        #pragma unroll
        for (int mt = 0; mt < 4; ++mt) {
            #pragma unroll
            for (int i = 0; i < 8; ++i)
                s += tanhf(acc[mt][i] + bc);
        }
        atomicAdd(&sAgg[ncol], s);        // ds_add_f32; lanes l and l+16 share ncol
    }
    __syncthreads();

    out[(long long)b * OUTW + CONCATW + t] = sAgg[t] * (1.0f / (float)P_);
}

// ---------------------------------------------------------------------------
extern "C" void kernel_launch(void* const* d_in, const int* in_sizes, int n_in,
                              void* d_out, int out_size, void* d_ws, size_t ws_size,
                              hipStream_t stream) {
    const float* feas    = (const float*)d_in[0];
    const float* workers = (const float*)d_in[1];
    const float* known   = (const float*)d_in[2];
    const float* unknown = (const float*)d_in[3];
    const float* differ  = (const float*)d_in[4];
    const float* qaturn  = (const float*)d_in[5];
    const int*   nodes   = (const int*)  d_in[6];
    const float* embed   = (const float*)d_in[7];
    const float* W       = (const float*)d_in[8];
    const float* bias    = (const float*)d_in[9];
    float*       out     = (float*)d_out;
    _Float16*    Wt      = (_Float16*)d_ws;   // 128 KB scratch for f16 W^T

    wt_convert_kernel<<<(E_ * E_) / 256, 256, 0, stream>>>(W, Wt);

    long long tot = (long long)B_ * CONCATW;
    concat_kernel<<<(unsigned)((tot + 255) / 256), 256, 0, stream>>>(
        feas, workers, known, unknown, differ, qaturn, out);

    gemm_agg_kernel<<<B_, 256, 0, stream>>>(nodes, embed, Wt, bias, out);
}
